// Quantizer_33071248179737
// MI455X (gfx1250) — matrix operand order repaired and checked by reading, not compiled, not run on hardware
//
#include <hip/hip_runtime.h>

typedef __attribute__((ext_vector_type(16))) __bf16 v16bf;
typedef __attribute__((ext_vector_type(8)))  __bf16 v8bf;
typedef __attribute__((ext_vector_type(8)))  float  v8f;

constexpr int D          = 128;          // latent dim
constexpr int NCODES     = 1024;         // codebook entries
constexpr int HW         = 4096;         // 64*64
constexpr int BATCH_STR  = D * HW;       // floats per batch image
constexpr int NROWS      = 32 * HW;      // 131072 latent vectors
constexpr int CHUNK      = 16;           // codes staged per iteration
constexpr int LDS_STRIDE = D + 8;        // bf16 elems per LDS row (bank pad)
constexpr int NCHUNKS    = NCODES / CHUNK;

// ---------------------------------------------------------------------------
// Phase 0: ||c||^2 for every codebook row
// ---------------------------------------------------------------------------
__global__ void cnorm_kernel(const float* __restrict__ cb, float* __restrict__ cnorm) {
  const int code = blockIdx.x * blockDim.x + threadIdx.x;   // 0..1023
  const float4* p = (const float4*)(cb + (size_t)code * D);
  float s = 0.f;
#pragma unroll
  for (int i = 0; i < D / 4; ++i) {
    float4 v = p[i];
    s += v.x * v.x + v.y * v.y + v.z * v.z + v.w * v.w;
  }
  cnorm[code] = s;
}

// ---------------------------------------------------------------------------
// Phase 1: fused GEMM (z . C^T via bf16 WMMA) + row argmin.
//   block = 256 threads (8 waves); each wave owns TWO 16-row A tiles (M=32).
//   Double-buffered LDS staging of 16 codebook rows, one barrier/iteration.
// ---------------------------------------------------------------------------
__global__ __launch_bounds__(256) void argmin_kernel(
    const float* __restrict__ z, const float* __restrict__ cb,
    const float* __restrict__ cnorm, int* __restrict__ idx_out) {
  __shared__ __align__(16) unsigned short Bsh[2][CHUNK * LDS_STRIDE];

  const int tid  = threadIdx.x;
  const int lane = tid & 31;
  const int wave = tid >> 5;
  const int hh   = lane >> 4;   // half-wave (0/1)
  const int col  = lane & 15;   // B/C column, also A row within a tile

  // ---- preload two A tiles (32 rows x 128 channels) in bf16 WMMA A-layout ----
  // A 16x32 bf16 operand, this code's contiguous k order (not AMD's table order): lane = row M; element j holds K = j + 16*hh
  v16bf A[2][4];
#pragma unroll
  for (int t = 0; t < 2; ++t) {
    const int n = blockIdx.x * 256 + wave * 32 + t * 16 + col;
    const float* zp = z + (size_t)(n >> 12) * BATCH_STR + (n & (HW - 1));
#pragma unroll
    for (int s = 0; s < 4; ++s)
#pragma unroll
      for (int j = 0; j < 16; ++j) {
        const int c = 32 * s + 16 * hh + j;
        A[t][s][j] = (__bf16)zp[(size_t)c * HW];
      }
  }

  float best[2][8];
  int   bidx[2][8];
#pragma unroll
  for (int t = 0; t < 2; ++t)
#pragma unroll
    for (int r = 0; r < 8; ++r) { best[t][r] = 3.4e38f; bidx[t][r] = 0; }

  const int srow = tid >> 4;         // staging: codebook row 0..15
  const int sc0  = (tid & 15) * 8;   // staging: first channel (8 floats/thread)

  // ---- prologue: stage chunk 0 into buffer 0 ----
  {
    const float4* gp = (const float4*)(cb + (size_t)srow * D + sc0);
    float4 v0 = gp[0], v1 = gp[1];
    v8bf pk;
    pk[0] = (__bf16)v0.x; pk[1] = (__bf16)v0.y; pk[2] = (__bf16)v0.z; pk[3] = (__bf16)v0.w;
    pk[4] = (__bf16)v1.x; pk[5] = (__bf16)v1.y; pk[6] = (__bf16)v1.z; pk[7] = (__bf16)v1.w;
    *(v8bf*)(&Bsh[0][srow * LDS_STRIDE + sc0]) = pk;
  }
  float cn = cnorm[col];
  __syncthreads();

  for (int chunk = 0; chunk < NCHUNKS; ++chunk) {
    const int cur       = chunk & 1;
    const int code_base = chunk * CHUNK;

    // ---- issue next-chunk global loads early (latency hidden by WMMAs) ----
    float4 nx0, nx1;
    float  cn_next = 0.f;
    const bool have_next = (chunk + 1) < NCHUNKS;
    if (have_next) {
      const float4* gp =
          (const float4*)(cb + (size_t)(code_base + CHUNK + srow) * D + sc0);
      nx0 = gp[0]; nx1 = gp[1];
      cn_next = cnorm[code_base + CHUNK + col];
      if (chunk + 2 < NCHUNKS)
        __builtin_prefetch(cb + (size_t)(code_base + 2 * CHUNK + srow) * D + sc0, 0, 1);
    }

    // ---- 32x16 dot tile: 8 x V_WMMA_F32_16X16X32_BF16, B shared by both tiles ----
    v8f acc0 = {}, acc1 = {};
#pragma unroll
    for (int s = 0; s < 4; ++s) {
      // B 32x16 operand, same contiguous k order as A (not AMD's table order): lane = column N; element j holds K = j + 16*hh
      const v8bf* bp = (const v8bf*)(&Bsh[cur][col * LDS_STRIDE + 32 * s + 16 * hh]);
      v8bf blo = bp[0], bhi = bp[1];
      v16bf bv = __builtin_shufflevector(blo, bhi,
                                         0, 1, 2, 3, 4, 5, 6, 7,
                                         8, 9, 10, 11, 12, 13, 14, 15);
      acc0 = __builtin_amdgcn_wmma_f32_16x16x32_bf16(false, A[0][s], false, bv,
                                                     (short)0, acc0, false, false);
      acc1 = __builtin_amdgcn_wmma_f32_16x16x32_bf16(false, A[1][s], false, bv,
                                                     (short)0, acc1, false, false);
    }

    // ---- dist = ||c||^2 - 2*dot  (||z||^2 constant per row, dropped) ----
#pragma unroll
    for (int r = 0; r < 8; ++r) {
      const float d0 = cn - 2.0f * acc0[r];
      if (d0 < best[0][r]) { best[0][r] = d0; bidx[0][r] = code_base + col; }
      const float d1 = cn - 2.0f * acc1[r];
      if (d1 < best[1][r]) { best[1][r] = d1; bidx[1][r] = code_base + col; }
    }

    // ---- convert + store next chunk into the other buffer ----
    if (have_next) {
      v8bf pk;
      pk[0] = (__bf16)nx0.x; pk[1] = (__bf16)nx0.y; pk[2] = (__bf16)nx0.z; pk[3] = (__bf16)nx0.w;
      pk[4] = (__bf16)nx1.x; pk[5] = (__bf16)nx1.y; pk[6] = (__bf16)nx1.z; pk[7] = (__bf16)nx1.w;
      *(v8bf*)(&Bsh[cur ^ 1][srow * LDS_STRIDE + sc0]) = pk;
      cn = cn_next;
    }
    __syncthreads();
  }

  // ---- min+index reduction across the 16 lanes of each half-wave ----
  // C/D layout: reg r on half hh is matrix row (r + 8*hh); columns live in lanes.
#pragma unroll
  for (int t = 0; t < 2; ++t) {
#pragma unroll
    for (int r = 0; r < 8; ++r) {
      float bv = best[t][r];
      int   bi = bidx[t][r];
#pragma unroll
      for (int off = 1; off < 16; off <<= 1) {
        float ov = __shfl_xor(bv, off, 32);
        int   oi = __shfl_xor(bi, off, 32);
        if (ov < bv || (ov == bv && oi < bi)) { bv = ov; bi = oi; }  // ties -> lower idx
      }
      best[t][r] = bv; bidx[t][r] = bi;
    }
    if (col == 0) {
      const int base = blockIdx.x * 256 + wave * 32 + t * 16 + 8 * hh;
#pragma unroll
      for (int r = 0; r < 8; ++r) idx_out[base + r] = bidx[t][r];
    }
  }
}

// ---------------------------------------------------------------------------
// Phase 2: quantized = codebook[idx], written back in (b, d, h, w) layout.
//   Coalesced stores along hw; codebook reads hit L2 (0.5 MB resident).
// ---------------------------------------------------------------------------
__global__ void gather_kernel(const float* __restrict__ cb, const int* __restrict__ idx,
                              float* __restrict__ out) {
  const int b  = blockIdx.y;
  const int hw = blockIdx.x * 32 + (threadIdx.x & 31);
  const int cg = threadIdx.x >> 5;            // 8 channel groups of 16
  const int row = idx[b * HW + hw];
  const float* src = cb + (size_t)row * D;
  float* dst = out + (size_t)b * BATCH_STR + hw;
#pragma unroll
  for (int k = 0; k < 16; ++k) {
    const int c = cg * 16 + k;
    dst[(size_t)c * HW] = src[c];
  }
}

// ---------------------------------------------------------------------------
extern "C" void kernel_launch(void* const* d_in, const int* in_sizes, int n_in,
                              void* d_out, int out_size, void* d_ws, size_t ws_size,
                              hipStream_t stream) {
  (void)in_sizes; (void)n_in; (void)out_size; (void)ws_size;
  const float* z  = (const float*)d_in[0];   // [32,128,64,64] fp32
  const float* cb = (const float*)d_in[1];   // [1024,128]     fp32
  float* out = (float*)d_out;                // [32,128,64,64] fp32

  float* cnorm = (float*)d_ws;                       // 1024 floats
  int*   idx   = (int*)((char*)d_ws + 4096);         // 131072 ints

  cnorm_kernel<<<NCODES / 128, 128, 0, stream>>>(cb, cnorm);
  argmin_kernel<<<NROWS / 256, 256, 0, stream>>>(z, cb, cnorm, idx);
  gather_kernel<<<dim3(HW / 32, 32), 256, 0, stream>>>(cb, idx, out);
}
